// SlotAttention_69879117906426
// MI455X (gfx1250) — matrix operand order repaired and checked by reading, not compiled, not run on hardware
//
#include <hip/hip_runtime.h>

// ---------------------------------------------------------------------------
// Slot Attention forward for MI455X (gfx1250), bf16 WMMA + f32 accumulate.
// B=32, N=4096, S=11 (padded to 16), D_IN=D=512, MLP=2048, ITERS=3.
// ---------------------------------------------------------------------------

typedef __attribute__((ext_vector_type(16))) __bf16        v16bf;
typedef __attribute__((ext_vector_type(8)))  float         v8f;
typedef __attribute__((ext_vector_type(4)))  unsigned int  v4u;

union F16U {
  v16bf v;
  v4u q[2];
  unsigned short s[16];
};

__device__ __forceinline__ unsigned short f2bf(float f) {
  unsigned int u = __builtin_bit_cast(unsigned int, f);
  u += 0x7fffu + ((u >> 16) & 1u);           // round-to-nearest-even
  return (unsigned short)(u >> 16);
}

__device__ __forceinline__ float wave_sum(float v) {
#pragma unroll
  for (int m = 16; m >= 1; m >>= 1) v += __shfl_xor(v, m, 32);
  return v;
}

__device__ __forceinline__ v8f zero8() {
  v8f z;
#pragma unroll
  for (int i = 0; i < 8; ++i) z[i] = 0.0f;
  return z;
}

__device__ __forceinline__ v8f wmma_bf16(v16bf a, v16bf b, v8f c) {
  // v_wmma_f32_16x16x32_bf16  (neg_a, A, neg_b, B, c_mod, C, reuse_a, reuse_b)
  return __builtin_amdgcn_wmma_f32_16x16x32_bf16(false, a, false, b, (short)0, c,
                                                 false, false);
}

// A fragment (16x32 bf16) from row-major [M][K] bf16 memory.
// lane: M = lane&15; 16 contiguous K at offset hf*16, hf = lane>>4.
__device__ __forceinline__ v16bf load_fragA(const unsigned short* p, int ld) {
  const int lane = threadIdx.x & 31;
  const unsigned short* r = p + (size_t)(lane & 15) * ld + ((lane >> 4) << 4);
  F16U f;
  f.q[0] = *(const v4u*)r;
  f.q[1] = *(const v4u*)(r + 8);
  return f.v;
}

// A fragment from row-major [M][K] f32 memory, scaled, converted to bf16.
__device__ __forceinline__ v16bf load_fragA_f32(const float* p, int ld, float scale) {
  const int lane = threadIdx.x & 31;
  const float* r = p + (size_t)(lane & 15) * ld + ((lane >> 4) << 4);
  F16U f;
#pragma unroll
  for (int c = 0; c < 2; ++c) {
    float4 a = *(const float4*)(r + c * 8);
    float4 b = *(const float4*)(r + c * 8 + 4);
    f.s[c * 8 + 0] = f2bf(a.x * scale);
    f.s[c * 8 + 1] = f2bf(a.y * scale);
    f.s[c * 8 + 2] = f2bf(a.z * scale);
    f.s[c * 8 + 3] = f2bf(a.w * scale);
    f.s[c * 8 + 4] = f2bf(b.x * scale);
    f.s[c * 8 + 5] = f2bf(b.y * scale);
    f.s[c * 8 + 6] = f2bf(b.z * scale);
    f.s[c * 8 + 7] = f2bf(b.w * scale);
  }
  return f.v;
}

// B fragment (32x16 bf16) via transpose-access of row-major [N][K] bf16:
// lane: N = lane&15; 16 contiguous K at offset hf*16.
__device__ __forceinline__ v16bf load_fragB_rows(const unsigned short* p, int ld) {
  const int lane = threadIdx.x & 31;
  const unsigned short* r = p + (size_t)(lane & 15) * ld + ((lane >> 4) << 4);
  F16U f;
  f.q[0] = *(const v4u*)r;
  f.q[1] = *(const v4u*)(r + 8);
  return f.v;
}

// B fragment from pre-packed fragment layout (one 32B contiguous read).
__device__ __forceinline__ v16bf load_fragB_packed(const unsigned short* frag,
                                                   int kt, int nt, int NT) {
  const int lane = threadIdx.x & 31;
  const unsigned short* p = frag + (((size_t)(kt * NT + nt) * 32) + lane) * 16;
  F16U f;
  f.q[0] = *(const v4u*)p;
  f.q[1] = *(const v4u*)(p + 8);
  return f.v;
}

// ---------------------------------------------------------------------------
// One-time weight preparation
// ---------------------------------------------------------------------------

// Row-major [K][Nfull] f32 -> bf16 fragment layout matching load_fragB_packed.
__global__ void pack_weight_kernel(const float* __restrict__ W,
                                   unsigned short* __restrict__ frag, int Nfull) {
  int id = blockIdx.x * 256 + threadIdx.x;
  int j = id & 15;
  int lane = (id >> 4) & 31;
  int tile = id >> 9;
  int NT = Nfull >> 4;
  int nt = tile % NT, kt = tile / NT;
  int k = kt * 32 + j + ((lane >> 4) << 4);
  int n = nt * 16 + (lane & 15);
  frag[id] = f2bf(W[(size_t)k * Nfull + n]);
}

__global__ void cvt_bf16_kernel(const float* __restrict__ src,
                                unsigned short* __restrict__ dst) {
  int id = blockIdx.x * 256 + threadIdx.x;
  dst[id] = f2bf(src[id]);
}

__global__ void init_slots_kernel(const float* __restrict__ slots_in,
                                  float* __restrict__ slots_cur) {
  int id = blockIdx.x * 256 + threadIdx.x;    // 512*512
  int p = id >> 9, d = id & 511;
  int b = p >> 4, s = p & 15;
  slots_cur[id] = (s < 11) ? slots_in[((size_t)b * 11 + s) * 512 + d] : 0.0f;
}

// ---------------------------------------------------------------------------
// K2: fused LN(inputs) + k/v projection.  32 rows per block.
//   k stored bf16 row-major [B*N][512]  (via LDS bounce, b128 stores)
//   v stored transposed bf16 [B][512][4096] (contiguous b128 stores)
// ---------------------------------------------------------------------------
#define LNP 528   // ushort stride for 512 cols (16B aligned, bank-staggered)

__global__ __launch_bounds__(256) void kv_kernel(
    const float* __restrict__ inputs, const float* __restrict__ g,
    const float* __restrict__ bvec, const unsigned short* __restrict__ WkF,
    const float* __restrict__ bk, const unsigned short* __restrict__ WvF,
    const float* __restrict__ bv, unsigned short* __restrict__ kbuf,
    unsigned short* __restrict__ vT) {
  __shared__ unsigned short xs[32 * LNP];
  __shared__ unsigned short bounce[8][16 * 24];
  const int wave = threadIdx.x >> 5, lane = threadIdx.x & 31;
  const size_t rowBase = (size_t)blockIdx.x * 32;

  // ---- LayerNorm phase: each wave normalizes 4 rows of 512 -> bf16 LDS ----
  for (int rr = 0; rr < 4; ++rr) {
    int rloc = wave * 4 + rr;
    const float* row = inputs + (rowBase + rloc) * 512;
    float s = 0.f, ss = 0.f;
    float4 vals[4];
#pragma unroll
    for (int p = 0; p < 4; ++p) {
      vals[p] = *(const float4*)(row + p * 128 + lane * 4);
      s += vals[p].x + vals[p].y + vals[p].z + vals[p].w;
      ss += vals[p].x * vals[p].x + vals[p].y * vals[p].y +
            vals[p].z * vals[p].z + vals[p].w * vals[p].w;
    }
    s = wave_sum(s);
    ss = wave_sum(ss);
    float mu = s * (1.0f / 512.0f);
    float rstd = rsqrtf(ss * (1.0f / 512.0f) - mu * mu + 1e-5f);
#pragma unroll
    for (int p = 0; p < 4; ++p) {
      int col = p * 128 + lane * 4;
      float4 gv = *(const float4*)(g + col);
      float4 bb = *(const float4*)(bvec + col);
      xs[rloc * LNP + col + 0] = f2bf((vals[p].x - mu) * rstd * gv.x + bb.x);
      xs[rloc * LNP + col + 1] = f2bf((vals[p].y - mu) * rstd * gv.y + bb.y);
      xs[rloc * LNP + col + 2] = f2bf((vals[p].z - mu) * rstd * gv.z + bb.z);
      xs[rloc * LNP + col + 3] = f2bf((vals[p].w - mu) * rstd * gv.w + bb.w);
    }
  }
  __syncthreads();

  // ---- GEMM phase: 2 mtiles x 32 ntiles x {k,v} = 128 tiles, 16 per wave ----
  for (int t = wave; t < 128; t += 8) {
    const int which = t >> 6;        // 0 = k, 1 = v
    const int mt = (t >> 5) & 1;
    const int nt = t & 31;
    v8f c = zero8();
    const unsigned short* A = xs + (size_t)mt * 16 * LNP;
    const unsigned short* WF = which ? WvF : WkF;
    for (int kt = 0; kt < 16; ++kt) {
      v16bf a = load_fragA(A + kt * 32, LNP);
      v16bf b = load_fragB_packed(WF, kt, nt, 32);
      c = wmma_bf16(a, b, c);
    }
    const int col = nt * 16 + (lane & 15);
    const int hf = lane >> 4;
    if (which == 0) {
      float bias = bk[col];
      unsigned short* tb = &bounce[wave][0];
#pragma unroll
      for (int r = 0; r < 8; ++r)
        tb[(size_t)(r + 8 * hf) * 24 + (lane & 15)] = f2bf(c[r] + bias);
      asm volatile("s_wait_dscnt 0" ::: "memory");
      int rloc2 = lane >> 1;                 // 0..15
      int cpart = (lane & 1) * 8;
      v4u d = *(const v4u*)&tb[(size_t)rloc2 * 24 + cpart];
      size_t grow = rowBase + mt * 16 + rloc2;
      *(v4u*)(kbuf + grow * 512 + nt * 16 + cpart) = d;
    } else {
      float bias = bv[col];
      size_t b_ = rowBase >> 12;                   // batch
      size_t nIn = (rowBase & 4095) + mt * 16 + 8 * hf;
      F16U pk;
#pragma unroll
      for (int r = 0; r < 8; ++r) pk.s[r] = f2bf(c[r] + bias);
      *(v4u*)(vT + ((b_ * 512 + col) * 4096) + nIn) = pk.q[0];
    }
  }
}

// ---------------------------------------------------------------------------
// K3: q = (LN(slots) @ Wq) * 1/sqrt(D), padded bf16 [B][16][512]
// ---------------------------------------------------------------------------
__global__ __launch_bounds__(256) void q_kernel(
    const float* __restrict__ slots_cur, const float* __restrict__ g,
    const float* __restrict__ bvec, const unsigned short* __restrict__ WqF,
    unsigned short* __restrict__ qpad) {
  __shared__ unsigned short qs[16 * LNP];
  const int b = blockIdx.x;
  const int wave = threadIdx.x >> 5, lane = threadIdx.x & 31;

  for (int h = 0; h < 2; ++h) {
    int row = wave + 8 * h;                   // 0..15
    if (row < 11) {
      const float* src = slots_cur + ((size_t)b * 16 + row) * 512;
      float s = 0.f, ss = 0.f;
      float4 vals[4];
#pragma unroll
      for (int p = 0; p < 4; ++p) {
        vals[p] = *(const float4*)(src + p * 128 + lane * 4);
        s += vals[p].x + vals[p].y + vals[p].z + vals[p].w;
        ss += vals[p].x * vals[p].x + vals[p].y * vals[p].y +
              vals[p].z * vals[p].z + vals[p].w * vals[p].w;
      }
      s = wave_sum(s);
      ss = wave_sum(ss);
      float mu = s * (1.0f / 512.0f);
      float rstd = rsqrtf(ss * (1.0f / 512.0f) - mu * mu + 1e-5f);
#pragma unroll
      for (int p = 0; p < 4; ++p) {
        int col = p * 128 + lane * 4;
        float4 gv = *(const float4*)(g + col);
        float4 bb = *(const float4*)(bvec + col);
        qs[row * LNP + col + 0] = f2bf((vals[p].x - mu) * rstd * gv.x + bb.x);
        qs[row * LNP + col + 1] = f2bf((vals[p].y - mu) * rstd * gv.y + bb.y);
        qs[row * LNP + col + 2] = f2bf((vals[p].z - mu) * rstd * gv.z + bb.z);
        qs[row * LNP + col + 3] = f2bf((vals[p].w - mu) * rstd * gv.w + bb.w);
      }
    } else {
#pragma unroll
      for (int p = 0; p < 4; ++p) {
        int col = p * 128 + lane * 4;
        qs[row * LNP + col + 0] = 0;
        qs[row * LNP + col + 1] = 0;
        qs[row * LNP + col + 2] = 0;
        qs[row * LNP + col + 3] = 0;
      }
    }
  }
  __syncthreads();

  const float qscale = 0.044194173824159216f;   // 1/sqrt(512)
  for (int i = 0; i < 4; ++i) {
    int nt = wave * 4 + i;
    v8f c = zero8();
    for (int kt = 0; kt < 16; ++kt) {
      v16bf a = load_fragA(qs + kt * 32, LNP);
      v16bf bfr = load_fragB_packed(WqF, kt, nt, 32);
      c = wmma_bf16(a, bfr, c);
    }
    int col = nt * 16 + (lane & 15);
    int hf = lane >> 4;
#pragma unroll
    for (int r = 0; r < 8; ++r) {
      int srow = r + 8 * hf;
      qpad[((size_t)b * 16 + srow) * 512 + col] = f2bf(c[r] * qscale);
    }
  }
}

// ---------------------------------------------------------------------------
// K4: logits = q @ k^T, softmax over the slot axis (in-register + shfl_xor 16)
//     attn (pre-renorm) -> f32 [B][16][4096]
// ---------------------------------------------------------------------------
__global__ __launch_bounds__(256) void attn_kernel(
    const unsigned short* __restrict__ qpad,
    const unsigned short* __restrict__ kbuf, float* __restrict__ attn) {
  const int b = blockIdx.y, chunk = blockIdx.x;
  const int wave = threadIdx.x >> 5, lane = threadIdx.x & 31;
  const unsigned short* qb = qpad + (size_t)b * 16 * 512;

  for (int i = 0; i < 2; ++i) {
    int tn = wave * 2 + i;                        // 0..15
    size_t nBase = (size_t)chunk * 256 + tn * 16;
    const unsigned short* kb = kbuf + ((size_t)b * 4096 + nBase) * 512;
    v8f c = zero8();
    for (int kt = 0; kt < 16; ++kt) {
      v16bf a = load_fragA(qb + kt * 32, 512);
      v16bf bb = load_fragB_rows(kb + kt * 32, 512);
      c = wmma_bf16(a, bb, c);
    }
    const int hf = lane >> 4;
    const size_t ncol = nBase + (lane & 15);
    float mx = -1e30f;
#pragma unroll
    for (int r = 0; r < 8; ++r) {
      int s = r + 8 * hf;
      if (s < 11) mx = fmaxf(mx, c[r]);
    }
    mx = fmaxf(mx, __shfl_xor(mx, 16, 32));
    float e[8];
    float sum = 0.f;
#pragma unroll
    for (int r = 0; r < 8; ++r) {
      int s = r + 8 * hf;
      e[r] = (s < 11) ? __expf(c[r] - mx) : 0.0f;
      sum += e[r];
    }
    sum += __shfl_xor(sum, 16, 32);
    float inv = 1.0f / sum;
#pragma unroll
    for (int r = 0; r < 8; ++r) {
      int s = r + 8 * hf;
      attn[((size_t)b * 16 + s) * 4096 + ncol] = e[r] * inv;
    }
  }
}

// Fused, deterministic renormalization: per padded row p,
//   attnbf[p][n] = bf16( attn[p][n] / (sum_n attn[p][n] + 1e-8) )
__global__ __launch_bounds__(256) void attn_norm_kernel(
    const float* __restrict__ attn, unsigned short* __restrict__ attnbf) {
  __shared__ float red[8];
  const int row = blockIdx.x;                 // 0..511
  const float* r = attn + (size_t)row * 4096;
  float2 vals[8];
  float s = 0.f;
#pragma unroll
  for (int p = 0; p < 8; ++p) {
    vals[p] = *(const float2*)(r + p * 512 + threadIdx.x * 2);
    s += vals[p].x + vals[p].y;
  }
  s = wave_sum(s);
  if ((threadIdx.x & 31) == 0) red[threadIdx.x >> 5] = s;
  __syncthreads();
  float tot = 0.f;
#pragma unroll
  for (int i = 0; i < 8; ++i) tot += red[i];
  float inv = 1.0f / (tot + 1e-8f);
#pragma unroll
  for (int p = 0; p < 8; ++p) {
    unsigned short lo = f2bf(vals[p].x * inv);
    unsigned short hi = f2bf(vals[p].y * inv);
    *(unsigned int*)(attnbf + (size_t)row * 4096 + p * 512 + threadIdx.x * 2) =
        (unsigned)lo | ((unsigned)hi << 16);
  }
}

// ---------------------------------------------------------------------------
// K5: updates = attnbf @ v (via vT), pure bf16 b128 fragment feeds.
//     Output stored bf16 (only consumer is the GRU GEMM A-operand).
// ---------------------------------------------------------------------------
__global__ __launch_bounds__(256) void updates_kernel(
    const unsigned short* __restrict__ attnbf, const unsigned short* __restrict__ vT,
    unsigned short* __restrict__ updates) {
  const int b = blockIdx.x;
  const int wave = threadIdx.x >> 5, lane = threadIdx.x & 31;
  const unsigned short* ab = attnbf + (size_t)b * 16 * 4096;
  for (int i = 0; i < 4; ++i) {
    int nt = wave * 4 + i;
    const unsigned short* vb = vT + ((size_t)b * 512 + nt * 16) * 4096;
    v8f c = zero8();
    for (int kt = 0; kt < 128; ++kt) {
      v16bf a = load_fragA(ab + kt * 32, 4096);
      v16bf bb = load_fragB_rows(vb + kt * 32, 4096);
      c = wmma_bf16(a, bb, c);
    }
    int col = nt * 16 + (lane & 15);
    int hf = lane >> 4;
#pragma unroll
    for (int r = 0; r < 8; ++r)
      updates[((size_t)b * 16 + r + 8 * hf) * 512 + col] = f2bf(c[r]);
  }
}

// ---------------------------------------------------------------------------
// K6a: GRU input/hidden GEMMs: gi = updates @ W_ih^T + b_ih, gh likewise
// ---------------------------------------------------------------------------
__global__ __launch_bounds__(256) void gru_gemm_kernel(
    const unsigned short* __restrict__ updates, const float* __restrict__ slots_cur,
    const unsigned short* __restrict__ WihB, const float* __restrict__ bih,
    const unsigned short* __restrict__ WhhB, const float* __restrict__ bhh,
    float* __restrict__ gi, float* __restrict__ gh) {
  const int mt = blockIdx.x;                          // 0..31
  const int wave = threadIdx.x >> 5, lane = threadIdx.x & 31;
  const int nt = blockIdx.y * 8 + wave;               // 0..95
  for (int which = 0; which < 2; ++which) {
    const unsigned short* W = which ? WhhB : WihB;
    const float* bias = which ? bhh : bih;
    float* out = which ? gh : gi;
    v8f c = zero8();
    for (int kt = 0; kt < 16; ++kt) {
      v16bf a = which
          ? load_fragA_f32(slots_cur + (size_t)mt * 16 * 512 + kt * 32, 512, 1.0f)
          : load_fragA(updates + (size_t)mt * 16 * 512 + kt * 32, 512);
      v16bf bb = load_fragB_rows(W + (size_t)nt * 16 * 512 + kt * 32, 512);
      c = wmma_bf16(a, bb, c);
    }
    int col = nt * 16 + (lane & 15);
    float bv_ = bias[col];
    int hf = lane >> 4;
#pragma unroll
    for (int r = 0; r < 8; ++r)
      out[((size_t)mt * 16 + r + 8 * hf) * 1536 + col] = c[r] + bv_;
  }
}

// K6b: GRU elementwise combine (torch gate order r,z,n)
__global__ void gru_elem_kernel(const float* __restrict__ gi,
                                const float* __restrict__ gh,
                                float* __restrict__ slots_cur) {
  int id = blockIdx.x * 256 + threadIdx.x;    // 512*512
  int p = id >> 9, d = id & 511;
  size_t base = (size_t)p * 1536 + d;
  float ir = gi[base], iz = gi[base + 512], inn = gi[base + 1024];
  float hr = gh[base], hz = gh[base + 512], hn = gh[base + 1024];
  float r = 1.0f / (1.0f + __expf(-(ir + hr)));
  float z = 1.0f / (1.0f + __expf(-(iz + hz)));
  float n = tanhf(inn + r * hn);
  float old = slots_cur[id];
  slots_cur[id] = (1.0f - z) * n + z * old;
}

// K6c: row LayerNorm -> bf16 (for MLP input)
__global__ __launch_bounds__(256) void ln_rows_kernel(
    const float* __restrict__ src, const float* __restrict__ g,
    const float* __restrict__ bvec, unsigned short* __restrict__ dstbf) {
  const int wave = threadIdx.x >> 5, lane = threadIdx.x & 31;
  const size_t row = (size_t)blockIdx.x * 8 + wave;
  const float* rp = src + row * 512;
  float s = 0.f, ss = 0.f;
  float4 vals[4];
#pragma unroll
  for (int p = 0; p < 4; ++p) {
    vals[p] = *(const float4*)(rp + p * 128 + lane * 4);
    s += vals[p].x + vals[p].y + vals[p].z + vals[p].w;
    ss += vals[p].x * vals[p].x + vals[p].y * vals[p].y +
          vals[p].z * vals[p].z + vals[p].w * vals[p].w;
  }
  s = wave_sum(s);
  ss = wave_sum(ss);
  float mu = s * (1.0f / 512.0f);
  float rstd = rsqrtf(ss * (1.0f / 512.0f) - mu * mu + 1e-5f);
#pragma unroll
  for (int p = 0; p < 4; ++p) {
    int col = p * 128 + lane * 4;
    float4 gv = *(const float4*)(g + col);
    float4 bb = *(const float4*)(bvec + col);
    unsigned short o0 = f2bf((vals[p].x - mu) * rstd * gv.x + bb.x);
    unsigned short o1 = f2bf((vals[p].y - mu) * rstd * gv.y + bb.y);
    unsigned short o2 = f2bf((vals[p].z - mu) * rstd * gv.z + bb.z);
    unsigned short o3 = f2bf((vals[p].w - mu) * rstd * gv.w + bb.w);
    uint2 pk;
    pk.x = (unsigned)o0 | ((unsigned)o1 << 16);
    pk.y = (unsigned)o2 | ((unsigned)o3 << 16);
    *(uint2*)(dstbf + row * 512 + col) = pk;
  }
}

// K6d: m1 = relu(hbf @ W1 + b1) -> bf16 [512][2048]
__global__ __launch_bounds__(256) void mlp1_kernel(
    const unsigned short* __restrict__ hbf, const unsigned short* __restrict__ W1F,
    const float* __restrict__ b1, unsigned short* __restrict__ m1) {
  const int mt = blockIdx.x;
  const int wave = threadIdx.x >> 5, lane = threadIdx.x & 31;
  const int nt = blockIdx.y * 8 + wave;               // 0..127
  v8f c = zero8();
  for (int kt = 0; kt < 16; ++kt) {
    v16bf a = load_fragA(hbf + (size_t)mt * 16 * 512 + kt * 32, 512);
    v16bf bb = load_fragB_packed(W1F, kt, nt, 128);
    c = wmma_bf16(a, bb, c);
  }
  int col = nt * 16 + (lane & 15);
  float bb_ = b1[col];
  int hf = lane >> 4;
#pragma unroll
  for (int r = 0; r < 8; ++r) {
    float vv = fmaxf(c[r] + bb_, 0.0f);
    m1[((size_t)mt * 16 + r + 8 * hf) * 2048 + col] = f2bf(vv);
  }
}

// K6e: slots = m1 @ W2 + b2 ; final iteration compacts padded rows into d_out
__global__ __launch_bounds__(256) void mlp2_kernel(
    const unsigned short* __restrict__ m1, const unsigned short* __restrict__ W2F,
    const float* __restrict__ b2, float* __restrict__ slots_cur,
    float* __restrict__ out_final, int final_iter) {
  const int mt = blockIdx.x;
  const int wave = threadIdx.x >> 5, lane = threadIdx.x & 31;
  const int nt = blockIdx.y * 8 + wave;               // 0..31
  v8f c = zero8();
  for (int kt = 0; kt < 64; ++kt) {
    v16bf a = load_fragA(m1 + (size_t)mt * 16 * 2048 + kt * 32, 2048);
    v16bf bb = load_fragB_packed(W2F, kt, nt, 32);
    c = wmma_bf16(a, bb, c);
  }
  int col = nt * 16 + (lane & 15);
  float bb_ = b2[col];
  int hf = lane >> 4;
#pragma unroll
  for (int r = 0; r < 8; ++r) {
    int p = mt * 16 + r + 8 * hf;
    float vv = c[r] + bb_;
    if (final_iter) {
      int b = p >> 4, s = p & 15;
      if (s < 11) out_final[((size_t)b * 11 + s) * 512 + col] = vv;
    } else {
      slots_cur[(size_t)p * 512 + col] = vv;
    }
  }
}

// ---------------------------------------------------------------------------
// Host launcher
// ---------------------------------------------------------------------------
extern "C" void kernel_launch(void* const* d_in, const int* in_sizes, int n_in,
                              void* d_out, int out_size, void* d_ws, size_t ws_size,
                              hipStream_t stream) {
  (void)in_sizes; (void)n_in; (void)out_size; (void)ws_size;
  const float* slots_in = (const float*)d_in[0];
  const float* inputs   = (const float*)d_in[1];
  const float* ln_in_g  = (const float*)d_in[2];
  const float* ln_in_b  = (const float*)d_in[3];
  const float* Wk       = (const float*)d_in[4];
  const float* bk       = (const float*)d_in[5];
  const float* Wv       = (const float*)d_in[6];
  const float* bv       = (const float*)d_in[7];
  const float* ln_q_g   = (const float*)d_in[8];
  const float* ln_q_b   = (const float*)d_in[9];
  const float* Wq       = (const float*)d_in[10];
  const float* W_ih     = (const float*)d_in[11];
  const float* b_ih     = (const float*)d_in[12];
  const float* W_hh     = (const float*)d_in[13];
  const float* b_hh     = (const float*)d_in[14];
  const float* ln_m_g   = (const float*)d_in[15];
  const float* ln_m_b   = (const float*)d_in[16];
  const float* W1       = (const float*)d_in[17];
  const float* b1       = (const float*)d_in[18];
  const float* W2       = (const float*)d_in[19];
  const float* b2       = (const float*)d_in[20];
  float* out = (float*)d_out;

  char* wsp = (char*)d_ws;
  auto alloc = [&](size_t bytes) -> char* {
    char* p = wsp;
    wsp += (bytes + 255) & ~(size_t)255;
    return p;
  };
  unsigned short* kbuf   = (unsigned short*)alloc((size_t)131072 * 512 * 2);
  unsigned short* vT     = (unsigned short*)alloc((size_t)32 * 512 * 4096 * 2);
  unsigned short* WkF    = (unsigned short*)alloc((size_t)512 * 512 * 2);
  unsigned short* WvF    = (unsigned short*)alloc((size_t)512 * 512 * 2);
  unsigned short* WqF    = (unsigned short*)alloc((size_t)512 * 512 * 2);
  unsigned short* W1F    = (unsigned short*)alloc((size_t)512 * 2048 * 2);
  unsigned short* W2F    = (unsigned short*)alloc((size_t)2048 * 512 * 2);
  unsigned short* WihB   = (unsigned short*)alloc((size_t)1536 * 512 * 2);
  unsigned short* WhhB   = (unsigned short*)alloc((size_t)1536 * 512 * 2);
  unsigned short* qpad   = (unsigned short*)alloc((size_t)32 * 16 * 512 * 2);
  float* attn            = (float*)alloc((size_t)32 * 16 * 4096 * 4);
  unsigned short* attnbf = (unsigned short*)alloc((size_t)32 * 16 * 4096 * 2);
  unsigned short* updates= (unsigned short*)alloc((size_t)512 * 512 * 2);
  float* gi              = (float*)alloc((size_t)512 * 1536 * 4);
  float* gh              = (float*)alloc((size_t)512 * 1536 * 4);
  float* slots_cur       = (float*)alloc((size_t)512 * 512 * 4);
  unsigned short* hbf    = (unsigned short*)alloc((size_t)512 * 512 * 2);
  unsigned short* m1     = (unsigned short*)alloc((size_t)512 * 2048 * 2);

  // --- one-time weight prep ---
  pack_weight_kernel<<<1024, 256, 0, stream>>>(Wk, WkF, 512);
  pack_weight_kernel<<<1024, 256, 0, stream>>>(Wv, WvF, 512);
  pack_weight_kernel<<<1024, 256, 0, stream>>>(Wq, WqF, 512);
  pack_weight_kernel<<<4096, 256, 0, stream>>>(W1, W1F, 2048);
  pack_weight_kernel<<<4096, 256, 0, stream>>>(W2, W2F, 512);
  cvt_bf16_kernel<<<3072, 256, 0, stream>>>(W_ih, WihB);
  cvt_bf16_kernel<<<3072, 256, 0, stream>>>(W_hh, WhhB);
  init_slots_kernel<<<1024, 256, 0, stream>>>(slots_in, slots_cur);

  // --- LN(inputs) + k/v projections (once) ---
  kv_kernel<<<4096, 256, 0, stream>>>(inputs, ln_in_g, ln_in_b, WkF, bk, WvF, bv,
                                      kbuf, vT);

  // --- slot attention iterations ---
  for (int it = 0; it < 3; ++it) {
    q_kernel<<<32, 256, 0, stream>>>(slots_cur, ln_q_g, ln_q_b, WqF, qpad);
    attn_kernel<<<dim3(16, 32), 256, 0, stream>>>(qpad, kbuf, attn);
    attn_norm_kernel<<<512, 256, 0, stream>>>(attn, attnbf);
    updates_kernel<<<32, 256, 0, stream>>>(attnbf, vT, updates);
    gru_gemm_kernel<<<dim3(32, 12), 256, 0, stream>>>(updates, slots_cur, WihB,
                                                      b_ih, WhhB, b_hh, gi, gh);
    gru_elem_kernel<<<1024, 256, 0, stream>>>(gi, gh, slots_cur);
    ln_rows_kernel<<<64, 256, 0, stream>>>(slots_cur, ln_m_g, ln_m_b, hbf);
    mlp1_kernel<<<dim3(32, 16), 256, 0, stream>>>(hbf, W1F, b1, m1);
    mlp2_kernel<<<dim3(32, 4), 256, 0, stream>>>(m1, W2F, b2, slots_cur, out,
                                                 (it == 2) ? 1 : 0);
  }
}
